// Quantizer_8126078124277
// MI455X (gfx1250) — matrix operand order repaired and checked by reading, not compiled, not run on hardware
//
#include <hip/hip_runtime.h>
#include <hip/hip_bf16.h>

// ---------------------------------------------------------------------------
// VQ-VAE quantizer for MI455X (gfx1250, wave32, WMMA).
//
// Score GEMM done as bf16x3 split-precision WMMA (hi*hi + lo*hi + hi*lo),
// ~fp32 accuracy, using v_wmma_f32_16x16x32_bf16. ||x||^2 is dropped since it
// does not affect the per-token argmin.
//
// B tiles are streamed into a double-buffered LDS tile with
// GLOBAL_LOAD_ASYNC_TO_LDS_B128 (ASYNCcnt) so the copy of code-chunk nc+1
// overlaps the 24 WMMAs of chunk nc.
//
// ws layout:
//   [0            , 524288)  Whi  bf16 [1024][256]
//   [524288       , 1048576) Wlo  bf16 [1024][256]
//   [1048576      , 1052672) wnorm f32 [1024]
//   [1052672      , 1183744) codeIdx i32 [32768]
//   [1183744      , 1183752) u64 fixed-point (2^20) sum of squared diffs
// ---------------------------------------------------------------------------

#ifndef USE_ASYNC_B
#define USE_ASYNC_B 1   // flip to 0 to restore the synchronous staging path
#endif

typedef __attribute__((ext_vector_type(16))) __bf16 v16bf;
typedef __attribute__((ext_vector_type(8)))  float  v8f;

union FragBF {
    v16bf v;
    uint4 q[2];
};

__device__ __forceinline__ unsigned short f2bf(float f) {
    unsigned int u = __float_as_uint(f);
    unsigned int r = u + 0x7FFFu + ((u >> 16) & 1u);   // round-to-nearest-even
    return (unsigned short)(r >> 16);
}
__device__ __forceinline__ float bf2f(unsigned short b) {
    return __uint_as_float(((unsigned int)b) << 16);
}

__device__ __forceinline__ void wait_async0() {
#if __has_builtin(__builtin_amdgcn_s_wait_asynccnt)
    __builtin_amdgcn_s_wait_asynccnt(0);
#else
    asm volatile("s_wait_asynccnt 0x0" ::: "memory");
#endif
}

// Issue 4 async 16B global->LDS copies: one 32B segment of the hi tile and
// one of the lo tile. IOFFSET is added to BOTH the global and LDS addresses,
// so offset:16 fetches the second half of each 32B segment.
__device__ __forceinline__ void issue_btile_copy(
    const unsigned short* __restrict__ gHi,   // uniform: Whi + nc*4096
    const unsigned short* __restrict__ gLo,   // uniform: Wlo + nc*4096
    unsigned dHi, unsigned dLo,               // per-lane LDS byte addresses
    unsigned voff)                            // per-lane global byte offset
{
    asm volatile(
        "global_load_async_to_lds_b128 %0, %4, %2\n\t"
        "global_load_async_to_lds_b128 %0, %4, %2 offset:16\n\t"
        "global_load_async_to_lds_b128 %1, %4, %3\n\t"
        "global_load_async_to_lds_b128 %1, %4, %3 offset:16"
        :
        : "v"(dHi), "v"(dLo), "s"(gHi), "s"(gLo), "v"(voff)
        : "memory");
}

// ---------------------------------------------------------------------------
// Kernel 1: codebook prep — bf16 hi/lo split + ||W_k||^2 + zero accumulator.
// ---------------------------------------------------------------------------
__global__ __launch_bounds__(256) void vq_prep(
    const float* __restrict__ W,
    unsigned short* __restrict__ Whi,
    unsigned short* __restrict__ Wlo,
    float* __restrict__ wnorm,
    unsigned long long* __restrict__ accum)
{
    const int k = blockIdx.x;
    const int t = threadIdx.x;
    const int idx = (k << 8) + t;

    float w = W[idx];
    unsigned short h = f2bf(w);
    Whi[idx] = h;
    Wlo[idx] = f2bf(w - bf2f(h));

    __shared__ float red[256];
    red[t] = w * w;
    __syncthreads();
    for (int s = 128; s > 0; s >>= 1) {
        if (t < s) red[t] += red[t + s];
        __syncthreads();
    }
    if (t == 0) wnorm[k] = red[0];
    if (k == 0 && t == 0) *accum = 0ull;   // re-zero every launch (graph-safe)
}

// ---------------------------------------------------------------------------
// Kernel 2: WMMA score GEMM + argmin.
// grid = 256 blocks * 128 tokens; block = 256 threads = 8 waves * 16 tokens.
// ---------------------------------------------------------------------------
__global__ __launch_bounds__(256) void vq_argmin(
    const float* __restrict__ x,
    const unsigned short* __restrict__ Whi,
    const unsigned short* __restrict__ Wlo,
    const float* __restrict__ wnorm,
    int* __restrict__ codeIdx)
{
    // Padded, 16B-aligned rows: A row = 40 bf16 (80B), B row = 264 bf16 (528B)
    __shared__ __align__(16) unsigned short sAhi[128 * 40];
    __shared__ __align__(16) unsigned short sAlo[128 * 40];
    __shared__ __align__(16) unsigned short sBhi[2][16 * 264];
    __shared__ __align__(16) unsigned short sBlo[2][16 * 264];

    const int tid  = threadIdx.x;
    const int lane = tid & 31;
    const int wv   = tid >> 5;       // wave id (owns tokens wv*16..wv*16+15)
    const int half = lane >> 4;      // lane group (0: lanes 0-15, 1: 16-31)
    const int nl   = lane & 15;      // row (A build) / column N (B, C/D)

    const int tok0  = blockIdx.x * 128;     // 128 | 1024 -> single batch/block
    const int batch = tok0 >> 10;
    const int hw0   = tok0 & 1023;

    // ---- Build A fragments in registers (ISA 16-bit A 16x32 layout:
    //      elems 0..7  = K[half*8 .. half*8+7],
    //      elems 8..15 = K[16+half*8 .. 16+half*8+7], M = lane%16) ----
    FragBF aHi[8], aLo[8];
    const int cl = tid >> 3;   // channel-local 0..31
    const int tg = tid & 7;    // token group of 16

    for (int kc = 0; kc < 8; ++kc) {
        const float4* src = reinterpret_cast<const float4*>(
            x + (((size_t)((batch << 8) + (kc << 5) + cl)) << 10) + hw0 + (tg << 4));
#pragma unroll
        for (int j = 0; j < 4; ++j) {
            float4 v4 = src[j];
            int tok = (tg << 4) + (j << 2);
            float vals[4] = {v4.x, v4.y, v4.z, v4.w};
#pragma unroll
            for (int e = 0; e < 4; ++e) {
                unsigned short h = f2bf(vals[e]);
                sAhi[(tok + e) * 40 + cl] = h;
                sAlo[(tok + e) * 40 + cl] = f2bf(vals[e] - bf2f(h));
            }
        }
        __syncthreads();
        const int row = (wv * 16 + nl) * 40;
        aHi[kc].q[0] = *reinterpret_cast<const uint4*>(&sAhi[row + half * 8]);
        aHi[kc].q[1] = *reinterpret_cast<const uint4*>(&sAhi[row + 16 + half * 8]);
        aLo[kc].q[0] = *reinterpret_cast<const uint4*>(&sAlo[row + half * 8]);
        aLo[kc].q[1] = *reinterpret_cast<const uint4*>(&sAlo[row + 16 + half * 8]);
        __syncthreads();
    }

    float minv[8];
    int   mini[8];
#pragma unroll
    for (int r = 0; r < 8; ++r) { minv[r] = 3.402823466e38f; mini[r] = 0; }

    const int codeL = tid >> 4;   // 0..15 (staging)
    const int seg   = tid & 15;   // 0..15 (staging)
    const int stageOff = codeL * 264 + seg * 16;      // ushort index in tile

#if USE_ASYNC_B
    const unsigned dHi0 = (unsigned)(size_t)&sBhi[0][stageOff];
    const unsigned dHi1 = (unsigned)(size_t)&sBhi[1][stageOff];
    const unsigned dLo0 = (unsigned)(size_t)&sBlo[0][stageOff];
    const unsigned dLo1 = (unsigned)(size_t)&sBlo[1][stageOff];
    const unsigned voff = (unsigned)(codeL * 512 + seg * 32);  // bytes in tile

    // Prologue: start copy of code-chunk 0 into buffer 0.
    issue_btile_copy(Whi, Wlo, dHi0, dLo0, voff);
#endif

#pragma unroll 1
    for (int nc = 0; nc < 64; ++nc) {
        const int buf = nc & 1;

#if USE_ASYNC_B
        wait_async0();            // own 4 copies for this buffer are done
        __syncthreads();          // everyone's copies done + prev compute done
        if (nc + 1 < 64) {        // overlap next chunk's copy with this compute
            const unsigned short* gHi = Whi + ((size_t)(nc + 1) << 12);
            const unsigned short* gLo = Wlo + ((size_t)(nc + 1) << 12);
            issue_btile_copy(gHi, gLo,
                             buf ? dHi0 : dHi1, buf ? dLo0 : dLo1, voff);
        }
#else
        {   // synchronous staging fallback (32B per thread per array)
            const size_t g = (((size_t)(nc * 16 + codeL)) << 8) + seg * 16;
            const uint4* gh = reinterpret_cast<const uint4*>(&Whi[g]);
            const uint4* gl = reinterpret_cast<const uint4*>(&Wlo[g]);
            uint4* dh = reinterpret_cast<uint4*>(&sBhi[buf][stageOff]);
            uint4* dl = reinterpret_cast<uint4*>(&sBlo[buf][stageOff]);
            dh[0] = gh[0]; dh[1] = gh[1];
            dl[0] = gl[0]; dl[1] = gl[1];
        }
        __syncthreads();
#endif

        v8f acc = {0.f, 0.f, 0.f, 0.f, 0.f, 0.f, 0.f, 0.f};
#pragma unroll
        for (int kc = 0; kc < 8; ++kc) {
            // 16-bit B 32x16 layout: elems 0..7 = K[half*8..+7], 8..15 = K[16+half*8..+7], N=lane%16
            FragBF bHi, bLo;
            const int bo = nl * 264 + kc * 32 + half * 8;
            bHi.q[0] = *reinterpret_cast<const uint4*>(&sBhi[buf][bo]);
            bHi.q[1] = *reinterpret_cast<const uint4*>(&sBhi[buf][bo + 16]);
            bLo.q[0] = *reinterpret_cast<const uint4*>(&sBlo[buf][bo]);
            bLo.q[1] = *reinterpret_cast<const uint4*>(&sBlo[buf][bo + 16]);
            acc = __builtin_amdgcn_wmma_f32_16x16x32_bf16(
                false, aHi[kc].v, false, bHi.v, (short)0, acc, false, false);
            acc = __builtin_amdgcn_wmma_f32_16x16x32_bf16(
                false, aLo[kc].v, false, bHi.v, (short)0, acc, false, false);
            acc = __builtin_amdgcn_wmma_f32_16x16x32_bf16(
                false, aHi[kc].v, false, bLo.v, (short)0, acc, false, false);
        }

        const float wn = wnorm[nc * 16 + nl];
#pragma unroll
        for (int r = 0; r < 8; ++r) {
            float s = wn - 2.0f * acc[r];       // ||x||^2 dropped (argmin-invariant)
            if (s < minv[r]) { minv[r] = s; mini[r] = nc * 16 + nl; }
        }
#if !USE_ASYNC_B
        __syncthreads();
#endif
    }

    // Cross-lane min+index reduce within each 16-lane half (C/D layout:
    // VGPR r holds row M = half*8 + r, column N = lane%16).
#pragma unroll
    for (int r = 0; r < 8; ++r) {
        float v = minv[r];
        int   i = mini[r];
#pragma unroll
        for (int m = 1; m < 16; m <<= 1) {
            float v2 = __shfl_xor(v, m, 32);
            int   i2 = __shfl_xor(i, m, 32);
            if (v2 < v || (v2 == v && i2 < i)) { v = v2; i = i2; }
        }
        if (nl == 0)
            codeIdx[tok0 + wv * 16 + half * 8 + r] = i;
    }
}

// ---------------------------------------------------------------------------
// Kernel 3: gather codebook rows -> NCHW output + squared-diff accumulation.
// ---------------------------------------------------------------------------
__global__ __launch_bounds__(256) void vq_out(
    const float* __restrict__ x,
    const float* __restrict__ W,
    const int* __restrict__ codeIdx,
    float* __restrict__ out,
    unsigned long long* __restrict__ accum)
{
    __shared__ float sQ[32 * 258];   // 32 tokens x 256 ch, pad 2 (bank-safe)
    __shared__ int   sIdx[32];
    __shared__ float red[256];

    const int tid = threadIdx.x;
    const int n0  = blockIdx.x * 32;
    const int b   = n0 >> 10;
    const int hw0 = n0 & 1023;

    if (tid < 32) sIdx[tid] = codeIdx[n0 + tid];
    __syncthreads();

#pragma unroll 1
    for (int r = 0; r < 32; ++r)           // coalesced 1KB row gather
        sQ[r * 258 + tid] = W[(sIdx[r] << 8) + tid];
    __syncthreads();

    float local = 0.f;
    const int wv = tid >> 5;
    const int t  = tid & 31;
#pragma unroll 1
    for (int i = 0; i < 32; ++i) {
        const int c = i * 8 + wv;
        const float q = sQ[t * 258 + c];
        const size_t addr = (((size_t)((b << 8) + c)) << 10) + hw0 + t;
        const float xi = x[addr];
        out[addr] = q;                      // 128B contiguous per wave
        const float d = q - xi;
        local += d * d;
    }

    red[tid] = local;
    __syncthreads();
    for (int s = 128; s > 0; s >>= 1) {
        if (tid < s) red[tid] += red[tid + s];
        __syncthreads();
    }
    if (tid == 0)   // fixed-point u64 atomic -> deterministic accumulation
        atomicAdd(accum, (unsigned long long)((double)red[0] * 1048576.0 + 0.5));
}

// ---------------------------------------------------------------------------
// Kernel 4: finalize the three scalar losses.
// ---------------------------------------------------------------------------
__global__ void vq_losses(const unsigned long long* __restrict__ accum,
                          float* __restrict__ out, int N)
{
    const double mse = ((double)*accum) / 1048576.0 / (double)N;
    out[N]     = (float)(1.25 * mse);   // vq_loss = 0.25*commit + emb
    out[N + 1] = (float)mse;            // embedding_loss
    out[N + 2] = (float)mse;            // commitment_loss
}

// ---------------------------------------------------------------------------
extern "C" void kernel_launch(void* const* d_in, const int* in_sizes, int n_in,
                              void* d_out, int out_size, void* d_ws, size_t ws_size,
                              hipStream_t stream)
{
    (void)in_sizes; (void)n_in; (void)ws_size;
    const float* x = (const float*)d_in[0];   // [32,256,32,32] f32
    const float* W = (const float*)d_in[1];   // [1024,256] f32
    float* out = (float*)d_out;
    const int N = out_size - 3;               // 8388608

    char* ws = (char*)d_ws;
    unsigned short* Whi = (unsigned short*)ws;
    unsigned short* Wlo = (unsigned short*)(ws + 524288);
    float* wnorm        = (float*)(ws + 1048576);
    int* codeIdx        = (int*)(ws + 1048576 + 4096);
    unsigned long long* accum =
        (unsigned long long*)(ws + 1048576 + 4096 + 131072);

    vq_prep  <<<1024, 256, 0, stream>>>(W, Whi, Wlo, wnorm, accum);
    vq_argmin<<< 256, 256, 0, stream>>>(x, Whi, Wlo, wnorm, codeIdx);
    vq_out   <<<1024, 256, 0, stream>>>(x, W, codeIdx, out, accum);
    vq_losses<<<   1,   1, 0, stream>>>(accum, out, N);
}
